// ReservoirLayer_18038862643552
// MI455X (gfx1250) — matrix operand order repaired and checked by reading, not compiled, not run on hardware
//
#include <hip/hip_runtime.h>

// ---------------------------------------------------------------------------
// CDNA5 (gfx1250) reservoir recurrence.
//   state tile (16 batches x 4ch x 16x16 + halo2) kept in LDS as bf16,
//   DOUBLE buffered (100 KB of the 320 KB WGP LDS) so each pixel-group's
//   result is written immediately after its WMMA chain -> 1 live accumulator,
//   no spills, 1 barrier per step. Weight fragments stay register-resident
//   across all 8 time steps (compiler hoists them; wave32 has 1024 VGPRs).
//   per pixel-group (4 adjacent x pixels): GEMM 16x16xK160 via 5x
//   v_wmma_f32_16x16x32_bf16, f32 accumulate,
//   leaky-relu = max(x, 0.1x) via raw v_max_num_f32 (2 VALU/elem).
// ---------------------------------------------------------------------------

typedef __bf16 bf16_t;
typedef __attribute__((ext_vector_type(16))) __bf16 v16bf;
typedef __attribute__((ext_vector_type(8)))  float  v8f;

#define N_GRID   16
#define CH       4
#define TSTEPS   8
#define KTOT     160          // 4 ch * 5 rows * 8 cols (union RF of 4 pixels)
#define NCHUNK   5            // 160 / 32
#define PADY     20           // 16 + 2*halo
#define PADX     20
#define CH_STRIDE    (PADY * PADX)           // 400 elems
#define BATCH_STRIDE (CH * CH_STRIDE)        // 1600 elems
#define LDS_ELEMS    (16 * BATCH_STRIDE)     // 25600 bf16 = 51200 B per buffer
#define WB_ELEMS     (64 * NCHUNK * 32 * 16) // 163840 bf16 in d_ws

union Frag {
    v16bf v;
    uint2 d2[4];
    uint4 d4[2];
};

// ---------------------------------------------------------------------------
// Repack W[16][16][4][4][5][5] f32 -> bf16 B-fragments.
// Group g = y*4 + xg covers pixels x = xg*4 + p, p=0..3.
// B matrix per group: [K=160][N=16], N = p*4 + o.
// k = c*40 + u*8 + j, where j = p + v  (column offset in the 8-wide RF union).
// Fragment layout (this code's contiguous k order, same as A): lane holds column N = lane&15;
// lanes 0-15 hold K = base+0..15, lanes 16-31 hold K = base+16..31 (chunk base
// = kc*32), 16 bf16 contiguous per lane.
// ---------------------------------------------------------------------------
__global__ void repack_weights(const float* __restrict__ W,
                               bf16_t* __restrict__ Wb) {
    int idx = blockIdx.x * blockDim.x + threadIdx.x;
    if (idx >= WB_ELEMS) return;
    int e    = idx & 15;
    int lane = (idx >> 4) & 31;
    int kc   = (idx >> 9) % NCHUNK;
    int g    = (idx >> 9) / NCHUNK;
    int y  = g >> 2;
    int xg = g & 3;
    int n  = lane & 15;
    int k  = kc * 32 + ((lane & 16) ? 16 : 0) + e;

    float val = 0.0f;
    if (k < KTOT) {
        int c  = k / 40;
        int r  = k % 40;
        int u  = r >> 3;
        int j  = r & 7;
        int p  = n >> 2;
        int o  = n & 3;
        int v  = j - p;
        if (v >= 0 && v < 5) {
            int x = xg * 4 + p;
            // W dims: [16][16][4][4][5][5] = [y][x][o][c][u][v]
            val = W[((((y * 16 + x) * 4 + o) * 4 + c) * 5 + u) * 5 + v];
        }
    }
    Wb[idx] = (bf16_t)val;
}

// ---------------------------------------------------------------------------
// Main kernel: one workgroup = 16-batch tile, 8 wave32.
// Wave w owns rows y = {2w, 2w+1}, each with 4 pixel-groups -> 8 groups/wave.
// ---------------------------------------------------------------------------
__global__ __launch_bounds__(256)
void reservoir_kernel(const float* __restrict__ xin_all,
                      const bf16_t* __restrict__ Wb,
                      float* __restrict__ out_all) {
    // double-buffered state: 2 x 51200 B = 100 KB (CDNA5: 320 KB LDS / WGP)
    __shared__ __attribute__((aligned(16))) bf16_t S[2][LDS_ELEMS];

    const int tid  = threadIdx.x;
    const int lane = tid & 31;
    const int wv   = tid >> 5;
    const int bt   = blockIdx.x;                 // batch tile (16 batches)

    // ---- zero both buffers (halos must stay zero forever) -----------------
    unsigned int* Su32 = (unsigned int*)&S[0][0];
    for (int i = tid; i < LDS_ELEMS; i += 256) Su32[i] = 0u;   // 2*ELEMS bf16
    __syncthreads();

    // ---- load/pad/reshape input into S[0] ---------------------------------
    const float* xin = xin_all + (size_t)bt * 16 * 784;
    for (int i = tid; i < 16 * 784; i += 256) {
        int b = i / 784;
        int f = i % 784;                         // f = c*256 + y*16 + x
        int c  = f >> 8;
        int yy = (f >> 4) & 15;
        int xx = f & 15;
        S[0][b * BATCH_STRIDE + c * CH_STRIDE + (yy + 2) * PADX + (xx + 2)] =
            (bf16_t)xin[i];
    }
    __syncthreads();

    const int m      = lane & 15;                // A row = local batch
    const int mrow0  = (lane & 16) ? 8 : 0;      // C/D row base for this lane
    const int ncol   = lane & 15;                // C/D column
    const int pp     = ncol >> 2;                // pixel-in-group
    const int oo     = ncol & 3;                 // out channel
    const int khalfa = (lane & 16) ? 16 : 0;     // A-fragment K sub-offset

    // ---- T recurrence steps ----------------------------------------------
    for (int t = 0; t < TSTEPS; ++t) {
        const bf16_t* __restrict__ Sr = &S[t & 1][0];
        bf16_t* __restrict__       Sw = &S[(t + 1) & 1][0];

        #pragma unroll
        for (int gi = 0; gi < 8; ++gi) {
            const int y  = wv * 2 + (gi >> 2);
            const int xg = gi & 3;
            const int g  = y * 4 + xg;
            const int x0 = xg * 4;

            v8f acc = {};
            #pragma unroll
            for (int kc = 0; kc < NCHUNK; ++kc) {
                // B fragment: 32 contiguous bytes per lane from repacked W
                // (loop-invariant across t -> compiler keeps these in VGPRs)
                Frag bf;
                const uint4* bp =
                    (const uint4*)(Wb + ((size_t)(g * NCHUNK + kc) * 32 + lane) * 16);
                bf.d4[0] = bp[0];
                bf.d4[1] = bp[1];

                // A fragment: two contiguous 8-element K runs from LDS state
                Frag af;
                const int base = kc * 32;
                const int k8a  = base + khalfa;       // run 1 (elems 0..7)
                const int k8b  = k8a + 8;             // run 2 (elems 8..15)
                {
                    int c0 = k8a / 40, u0 = (k8a % 40) >> 3;
                    const uint2* ap = (const uint2*)
                        &Sr[m * BATCH_STRIDE + c0 * CH_STRIDE + (y + u0) * PADX + x0];
                    af.d2[0] = ap[0];
                    af.d2[1] = ap[1];
                }
                {
                    int c1 = k8b / 40, u1 = (k8b % 40) >> 3;
                    const uint2* ap = (const uint2*)
                        &Sr[m * BATCH_STRIDE + c1 * CH_STRIDE + (y + u1) * PADX + x0];
                    af.d2[2] = ap[0];
                    af.d2[3] = ap[1];
                }

                acc = __builtin_amdgcn_wmma_f32_16x16x32_bf16(
                    /*neg_a=*/false, af.v, /*neg_b=*/false, bf.v,
                    /*c_mod=*/(short)0, acc,
                    /*reuse_a=*/false, /*reuse_b=*/false);
            }

            // leaky-relu: slope 0.1 < 1  =>  exactly max(x, 0.1x).
            // raw v_max_num_f32 avoids clang's NaN-canonicalize self-max,
            // pinning this to v_mul + v_max (2 VALU/elem).
            const int xx = x0 + pp;
            #pragma unroll
            for (int i = 0; i < 8; ++i) {
                float d = acc[i];
                float s = 0.1f * d;
                asm("v_max_num_f32 %0, %1, %2" : "=v"(d) : "v"(d), "v"(s));
                const int mm = mrow0 + i;
                Sw[mm * BATCH_STRIDE + oo * CH_STRIDE + (y + 2) * PADX + (xx + 2)] =
                    (bf16_t)d;
            }
        }

        __syncthreads();   // next state fully written; prev reads done
    }

    // ---- channel-mean pooling from S[0] -> out [16][256] f32 --------------
    float* outp = out_all + (size_t)bt * 16 * 256;
    for (int i = tid; i < 16 * 256; i += 256) {
        int b   = i >> 8;
        int rem = i & 255;
        int yy  = rem >> 4;
        int xx  = rem & 15;
        float s = 0.0f;
        #pragma unroll
        for (int c = 0; c < CH; ++c)
            s += (float)S[0][b * BATCH_STRIDE + c * CH_STRIDE + (yy + 2) * PADX + (xx + 2)];
        outp[i] = s * 0.25f;
    }
}

// ---------------------------------------------------------------------------
extern "C" void kernel_launch(void* const* d_in, const int* in_sizes, int n_in,
                              void* d_out, int out_size, void* d_ws, size_t ws_size,
                              hipStream_t stream) {
    const float* x = (const float*)d_in[0];   // [8192, 784] f32
    const float* W = (const float*)d_in[1];   // [16,16,4,4,5,5] f32
    float* out     = (float*)d_out;           // [8192, 256] f32
    bf16_t* Wb     = (bf16_t*)d_ws;           // 163840 bf16 = 320 KB scratch

    // 1) repack weights into WMMA B-fragment layout
    {
        int threads = 256;
        int blocks  = (WB_ELEMS + threads - 1) / threads;   // 640
        repack_weights<<<blocks, threads, 0, stream>>>(W, Wb);
    }
    // 2) recurrence: 8192 batches / 16 per workgroup = 512 workgroups
    {
        reservoir_kernel<<<512, 256, 0, stream>>>(x, Wb, out);
    }
}
